// _HFStreamingLLMWrapper_44890998177949
// MI455X (gfx1250) — matrix operand order repaired and checked by reading, not compiled, not run on hardware
//
#include <hip/hip_runtime.h>
#include <hip/hip_bf16.h>
#include <cstdint>

// ---------------------------------------------------------------------------
// GPT-2 attention block w/ StreamingLLM sink+window mask on gfx1250 (MI455X)
//   B=1, S=4096, D=768, H=12, hd=64, sink=4, window=256
// Pipeline:
//   [f32->f16 convert + weight transpose]  (one-time, bandwidth-trivial)
//   [WMMA f16 GEMM qkv, async-to-LDS double buffered]
//   [windowed flash-attn, WMMA]
//   [WMMA GEMM proj, async-to-LDS double buffered]
// ---------------------------------------------------------------------------

typedef __attribute__((ext_vector_type(16))) _Float16 v16h;
typedef __attribute__((ext_vector_type(8)))  _Float16 v8h;
typedef __attribute__((ext_vector_type(4)))  _Float16 v4h;
typedef __attribute__((ext_vector_type(8)))  float    v8f;
typedef __attribute__((ext_vector_type(4)))  float    v4f;

#define SLEN 4096
#define DMODEL 768
#define NH 12
#define HD 64

// WMMA D = A*B + C,  16x16x32 f16 -> f32
__device__ __forceinline__ v8f wmma_f16(v16h a, v16h b, v8f c) {
  return __builtin_amdgcn_wmma_f32_16x16x32_f16(
      /*neg_a=*/false, a, /*neg_b=*/false, b,
      /*c_mod=*/(short)0, c, /*reuse_a=*/false, /*reuse_b=*/false);
}

// Async 16B copy global -> LDS (ASYNCcnt-tracked).  LDS byte address is the
// low 32 bits of the generic pointer (LDS aperture: addr[31:0] = LDS offset).
__device__ __forceinline__ void async_b128(void* lds, const void* g) {
  asm volatile("global_load_async_to_lds_b128 %0, %1, off"
               :: "v"((uint32_t)(uintptr_t)lds), "v"((uint64_t)(uintptr_t)g)
               : "memory");
}
__device__ __forceinline__ void wait_async0() {
  asm volatile("s_wait_asynccnt 0x0" ::: "memory");
}

// A-fragment (16x32 f16, MxK). Per ISA 7.12.2:
//   lanes 0-15: M=lane,    K = {0..7, 16..23}
//   lanes16-31: M=lane-16, K = {8..15, 24..31}
__device__ __forceinline__ v16h ldA(const _Float16* base, int lda) {
  const int lid = threadIdx.x & 31;
  const int row = lid & 15;
  const int kg  = (lid >> 4) << 3;          // 0 or 8
  const v8h* p0 = reinterpret_cast<const v8h*>(base + row * lda + kg);
  const v8h* p1 = reinterpret_cast<const v8h*>(base + row * lda + 16 + kg);
  v8h lo = *p0, hi = *p1;
  return __builtin_shufflevector(lo, hi, 0,1,2,3,4,5,6,7,8,9,10,11,12,13,14,15);
}

// B-fragment (32x16 f16, KxN), source stored COLUMN-major as Bt[n][k] with
// row stride ldb.  lanes 0-15 hold K={0..7,16..23} (col=lane), lanes16-31 K={8..15,24..31}.
__device__ __forceinline__ v16h ldB(const _Float16* baseT, int ldb) {
  const int lid = threadIdx.x & 31;
  const int col = lid & 15;
  const int kh  = (lid >> 4) << 3;          // 0 or 8
  const v8h* p = reinterpret_cast<const v8h*>(baseT + col * ldb + kh);
  v8h lo = p[0], hi = p[2];
  return __builtin_shufflevector(lo, hi, 0,1,2,3,4,5,6,7,8,9,10,11,12,13,14,15);
}

// ---------------------------------------------------------------------------
// One-time converts: f32 -> f16 elementwise, and f32 -> f16 transposed.
// ---------------------------------------------------------------------------
__global__ void __launch_bounds__(256)
k_cvt(const float* __restrict__ src, _Float16* __restrict__ dst, int n4) {
  int i = blockIdx.x * 256 + threadIdx.x;
  if (i < n4) {
    v4f v = reinterpret_cast<const v4f*>(src)[i];
    v4h h;
    h[0] = (_Float16)v[0]; h[1] = (_Float16)v[1];
    h[2] = (_Float16)v[2]; h[3] = (_Float16)v[3];
    reinterpret_cast<v4h*>(dst)[i] = h;
  }
}

// dst[c*R + r] = (f16) src[r*C + c]   (R rows, C cols; R,C multiples of 32)
__global__ void __launch_bounds__(256)
k_tr_cvt(const float* __restrict__ src, _Float16* __restrict__ dst, int R, int C) {
  __shared__ float tile[32][33];
  const int bx = blockIdx.x * 32;           // C dim
  const int by = blockIdx.y * 32;           // R dim
  const int tx = threadIdx.x & 31, ty = threadIdx.x >> 5;
#pragma unroll
  for (int j = 0; j < 32; j += 8)
    tile[ty + j][tx] = src[(size_t)(by + ty + j) * C + bx + tx];
  __syncthreads();
#pragma unroll
  for (int j = 0; j < 32; j += 8)
    dst[(size_t)(bx + ty + j) * R + by + tx] = (_Float16)tile[tx][ty + j];
}

// ---------------------------------------------------------------------------
// GEMM 1: qkv = Xh[4096,768] @ WaT^T + b ; scatter to f16 Q,K,VT.
// A,B f16 in global; tiles streamed with global_load_async_to_lds_b128,
// double buffered.  Block tile 128x128, 8 waves (4Mx2N), K-step 32.
// ---------------------------------------------------------------------------
#define BM 128
#define BN 128
#define BK 32
#define LDT 40            // padded LDS stride (halves): 80B rows, conflict-free
#define TBUF (BM * LDT)   // halves per buffer

__global__ void __launch_bounds__(256)
k_gemm_qkv(const _Float16* __restrict__ Ah, const _Float16* __restrict__ WT,
           const float* __restrict__ bias,
           _Float16* __restrict__ Qh, _Float16* __restrict__ Kh,
           _Float16* __restrict__ VT)
{
  __shared__ _Float16 As[2 * TBUF];
  __shared__ _Float16 Bs[2 * TBUF];
  const int t   = threadIdx.x;
  const int lid = t & 31;
  const int w   = t >> 5;
  const int wm  = w >> 1;           // 0..3 -> 32-row slice
  const int wn  = w & 1;            // 0..1 -> 64-col slice
  const int m0  = blockIdx.y * BM;
  const int n0  = blockIdx.x * BN;
  const int hi  = lid >> 4, lo = lid & 15;

  v8f acc[2][4] = {};

  // issue one 128x32 f16 tile for A and B into buffer `buf` (4 B128s/thread)
  auto issue = [&](int k0, int buf) {
#pragma unroll
    for (int it = 0; it < 2; ++it) {
      int c = t + it * 256;                 // 0..511: row = c>>2, 16B chunk q
      int row = c >> 2, q = (c & 3) * 8;
      async_b128(&As[buf * TBUF + row * LDT + q],
                 Ah + (size_t)(m0 + row) * DMODEL + k0 + q);
      async_b128(&Bs[buf * TBUF + row * LDT + q],
                 WT + (size_t)(n0 + row) * DMODEL + k0 + q);
    }
  };

  issue(0, 0);
  wait_async0();
  __syncthreads();

  int buf = 0;
  for (int k0 = 0; k0 < DMODEL; k0 += BK, buf ^= 1) {
    if (k0 + BK < DMODEL) issue(k0 + BK, buf ^ 1);

    v16h a0 = ldA(&As[buf * TBUF + (wm * 32 + 0)  * LDT], LDT);
    v16h a1 = ldA(&As[buf * TBUF + (wm * 32 + 16) * LDT], LDT);
#pragma unroll
    for (int ni = 0; ni < 4; ++ni) {
      v16h b = ldB(&Bs[buf * TBUF + (wn * 64 + ni * 16) * LDT], LDT);
      acc[0][ni] = wmma_f16(a0, b, acc[0][ni]);
      acc[1][ni] = wmma_f16(a1, b, acc[1][ni]);
    }
    wait_async0();
    __syncthreads();
  }

  // Epilogue: bias, split q/k/v, scale q by 1/sqrt(64), scatter f16.
#pragma unroll
  for (int mi = 0; mi < 2; ++mi)
#pragma unroll
    for (int ni = 0; ni < 4; ++ni)
#pragma unroll
      for (int r = 0; r < 8; ++r) {
        int s  = m0 + wm * 32 + mi * 16 + r + (hi << 3);
        int cN = n0 + wn * 64 + ni * 16 + lo;
        float v = acc[mi][ni][r] + bias[cN];
        int which = cN / DMODEL;
        int dd = cN - which * DMODEL;
        int h = dd >> 6, d = dd & 63;
        if (which == 0)
          Qh[((size_t)h * SLEN + s) * HD + d] = (_Float16)(v * 0.125f);
        else if (which == 1)
          Kh[((size_t)h * SLEN + s) * HD + d] = (_Float16)v;
        else
          VT[((size_t)h * HD + d) * SLEN + s] = (_Float16)v;
      }
}

// ---------------------------------------------------------------------------
// Windowed flash attention: one wave per (head, 16-query tile).
// Visits only the sink chunk + window-covering 32-key chunks.
// ---------------------------------------------------------------------------
#define PLD 40

__global__ void __launch_bounds__(256)
k_attn(const _Float16* __restrict__ Qh, const _Float16* __restrict__ Kh,
       const _Float16* __restrict__ VT, _Float16* __restrict__ CtxH,
       const int* __restrict__ d_sink, const int* __restrict__ d_win)
{
  __shared__ _Float16 Pl[8][16 * PLD];
  const int t   = threadIdx.x;
  const int lid = t & 31;
  const int w   = t >> 5;
  const int tile = blockIdx.x * 8 + w;
  const int h   = tile >> 8;                // 256 query tiles per head
  const int q0  = (tile & 255) << 4;
  const int hi  = lid >> 4, lo = lid & 15;
  const int sink = d_sink[0];               // 4   (assumed <= 32)
  const int win  = d_win[0];                // 256
  _Float16* P = Pl[w];

  const _Float16* qbase = Qh + ((size_t)h * SLEN + q0) * HD;
  v16h qa0 = ldA(qbase, HD);
  v16h qa1 = ldA(qbase + 32, HD);

  v8f o[4] = {};
  float m[8], l[8];
#pragma unroll
  for (int r = 0; r < 8; ++r) { m[r] = -1e30f; l[r] = 0.f; }

  int low = q0 - win + 1;
  int c0 = (low >= 32) ? (low >> 5) : 1;
  int c1 = (q0 + 15) >> 5;

  for (int ci = c0 - 1; ci <= c1; ++ci) {
    const int kv0 = (ci < c0) ? 0 : (ci << 5);   // ci==c0-1 -> sink chunk

    // scores S = Q K^T for 32 keys (two 16-key n-tiles)
    v8f sc[2];
#pragma unroll
    for (int nt = 0; nt < 2; ++nt) {
      const _Float16* kb = Kh + ((size_t)h * SLEN + kv0 + nt * 16) * HD;
      v8f z = {};
      z = wmma_f16(qa0, ldB(kb, HD), z);        // hd 0..31
      z = wmma_f16(qa1, ldB(kb + 32, HD), z);   // hd 32..63
      sc[nt] = z;
    }

    // mask: visible iff j<=i && (j<sink || j>=i-win+1)
#pragma unroll
    for (int nt = 0; nt < 2; ++nt)
#pragma unroll
      for (int r = 0; r < 8; ++r) {
        int i = q0 + r + (hi << 3);
        int j = kv0 + nt * 16 + lo;
        bool vis = (j <= i) && ((j < sink) || (j >= i - win + 1));
        if (!vis) sc[nt][r] = -1e30f;
      }

    // online softmax (reduce across the 16-lane half that owns the row)
#pragma unroll
    for (int r = 0; r < 8; ++r) {
      float mx = fmaxf(sc[0][r], sc[1][r]);
#pragma unroll
      for (int d = 1; d < 16; d <<= 1) mx = fmaxf(mx, __shfl_xor(mx, d));
      float mn = fmaxf(m[r], mx);
      float al = __expf(m[r] - mn);
      float p0 = __expf(sc[0][r] - mn);
      float p1 = __expf(sc[1][r] - mn);
      sc[0][r] = p0; sc[1][r] = p1;
      float sm = p0 + p1;
#pragma unroll
      for (int d = 1; d < 16; d <<= 1) sm += __shfl_xor(sm, d);
      l[r] = l[r] * al + sm;
      m[r] = mn;
#pragma unroll
      for (int nt = 0; nt < 4; ++nt) o[nt][r] *= al;
    }

    // P (C-layout f32) -> LDS f16 -> reload as A-fragment
#pragma unroll
    for (int nt = 0; nt < 2; ++nt)
#pragma unroll
      for (int r = 0; r < 8; ++r)
        P[(r + (hi << 3)) * PLD + nt * 16 + lo] = (_Float16)sc[nt][r];
    __asm volatile("" ::: "memory");   // DS ops are in-order per wave
    v16h pa = ldA(P, PLD);

    // O += P @ V   (VT[h][d][s]: key index contiguous -> B frag)
#pragma unroll
    for (int nt = 0; nt < 4; ++nt) {
      const _Float16* vb = VT + ((size_t)h * HD + nt * 16) * SLEN + kv0;
      o[nt] = wmma_f16(pa, ldB(vb, SLEN), o[nt]);
    }
  }

  // finalize: divide by l, store ctx f16 [S][768]
#pragma unroll
  for (int r = 0; r < 8; ++r) {
    float inv = 1.0f / l[r];
#pragma unroll
    for (int nt = 0; nt < 4; ++nt) o[nt][r] *= inv;
  }
#pragma unroll
  for (int nt = 0; nt < 4; ++nt)
#pragma unroll
    for (int r = 0; r < 8; ++r)
      CtxH[(size_t)(q0 + r + (hi << 3)) * DMODEL + h * HD + nt * 16 + lo] =
          (_Float16)o[nt][r];
}

// ---------------------------------------------------------------------------
// GEMM 2: out = CtxH[4096,768] @ WpT^T + b  (fp32 out), async double-buffered.
// ---------------------------------------------------------------------------
__global__ void __launch_bounds__(256)
k_gemm_proj(const _Float16* __restrict__ Ah, const _Float16* __restrict__ WT,
            const float* __restrict__ bias, float* __restrict__ out)
{
  __shared__ _Float16 As[2 * TBUF];
  __shared__ _Float16 Bs[2 * TBUF];
  const int t   = threadIdx.x;
  const int lid = t & 31;
  const int w   = t >> 5;
  const int wm  = w >> 1;
  const int wn  = w & 1;
  const int m0  = blockIdx.y * BM;
  const int n0  = blockIdx.x * BN;
  const int hi  = lid >> 4, lo = lid & 15;

  v8f acc[2][4] = {};

  auto issue = [&](int k0, int buf) {
#pragma unroll
    for (int it = 0; it < 2; ++it) {
      int c = t + it * 256;
      int row = c >> 2, q = (c & 3) * 8;
      async_b128(&As[buf * TBUF + row * LDT + q],
                 Ah + (size_t)(m0 + row) * DMODEL + k0 + q);
      async_b128(&Bs[buf * TBUF + row * LDT + q],
                 WT + (size_t)(n0 + row) * DMODEL + k0 + q);
    }
  };

  issue(0, 0);
  wait_async0();
  __syncthreads();

  int buf = 0;
  for (int k0 = 0; k0 < DMODEL; k0 += BK, buf ^= 1) {
    if (k0 + BK < DMODEL) issue(k0 + BK, buf ^ 1);

    v16h a0 = ldA(&As[buf * TBUF + (wm * 32 + 0)  * LDT], LDT);
    v16h a1 = ldA(&As[buf * TBUF + (wm * 32 + 16) * LDT], LDT);
#pragma unroll
    for (int ni = 0; ni < 4; ++ni) {
      v16h b = ldB(&Bs[buf * TBUF + (wn * 64 + ni * 16) * LDT], LDT);
      acc[0][ni] = wmma_f16(a0, b, acc[0][ni]);
      acc[1][ni] = wmma_f16(a1, b, acc[1][ni]);
    }
    wait_async0();
    __syncthreads();
  }

#pragma unroll
  for (int mi = 0; mi < 2; ++mi)
#pragma unroll
    for (int ni = 0; ni < 4; ++ni)
#pragma unroll
      for (int r = 0; r < 8; ++r) {
        int s  = m0 + wm * 32 + mi * 16 + r + (hi << 3);
        int cN = n0 + wn * 64 + ni * 16 + lo;
        out[(size_t)s * DMODEL + cN] = acc[mi][ni][r] + bias[cN];
      }
}

// ---------------------------------------------------------------------------
extern "C" void kernel_launch(void* const* d_in, const int* in_sizes, int n_in,
                              void* d_out, int out_size, void* d_ws, size_t ws_size,
                              hipStream_t stream) {
  const float* X  = (const float*)d_in[0];   // hidden_states [1,4096,768]
  const float* Wa = (const float*)d_in[1];   // w_attn [768,2304]
  const float* Ba = (const float*)d_in[2];   // b_attn [2304]
  const float* Wp = (const float*)d_in[3];   // w_proj [768,768]
  const float* Bp = (const float*)d_in[4];   // b_proj [768]
  const int* sink = (const int*)d_in[5];
  const int* win  = (const int*)d_in[6];
  float* out = (float*)d_out;

  // Workspace carve-out (~36 MB total)
  char* ws = (char*)d_ws;
  size_t off = 0;
  auto carve = [&](size_t bytes) {
    void* p = ws + off;
    off += (bytes + 255) & ~(size_t)255;
    return p;
  };
  _Float16* Xh  = (_Float16*)carve((size_t)SLEN * DMODEL * 2);     // f16 X
  _Float16* WaT = (_Float16*)carve((size_t)2304 * DMODEL * 2);     // f16 Wa^T
  _Float16* WpT = (_Float16*)carve((size_t)DMODEL * DMODEL * 2);   // f16 Wp^T
  _Float16* Qh  = (_Float16*)carve((size_t)NH * SLEN * HD * 2);    // [h][s][d]
  _Float16* Kh  = (_Float16*)carve((size_t)NH * SLEN * HD * 2);    // [h][s][d]
  _Float16* VT  = (_Float16*)carve((size_t)NH * HD * SLEN * 2);    // [h][d][s]
  _Float16* Cx  = (_Float16*)carve((size_t)SLEN * DMODEL * 2);     // ctx f16
  (void)ws_size; (void)in_sizes; (void)n_in; (void)out_size;

  k_cvt<<<dim3(SLEN * DMODEL / 4 / 256), 256, 0, stream>>>(X, Xh, SLEN * DMODEL / 4);
  k_tr_cvt<<<dim3(2304 / 32, DMODEL / 32), 256, 0, stream>>>(Wa, WaT, DMODEL, 2304);
  k_tr_cvt<<<dim3(DMODEL / 32, DMODEL / 32), 256, 0, stream>>>(Wp, WpT, DMODEL, DMODEL);

  k_gemm_qkv<<<dim3(2304 / BN, SLEN / BM), 256, 0, stream>>>(Xh, WaT, Ba, Qh, Kh, VT);
  k_attn<<<dim3((NH * (SLEN / 16)) / 8), 256, 0, stream>>>(Qh, Kh, VT, Cx, sink, win);
  k_gemm_proj<<<dim3(DMODEL / BN, SLEN / BM), 256, 0, stream>>>(Cx, WpT, Bp, out);
}
